// MultiTaskGNN_45397804319447
// MI455X (gfx1250) — matrix operand order repaired and checked by reading, not compiled, not run on hardware
//
#include <hip/hip_runtime.h>
#include <hip/hip_bf16.h>

// ---------------------------------------------------------------------------
// Sizes (match reference)
// ---------------------------------------------------------------------------
#define N_NODES  50000
#define N_EDGES  600000
#define N_GRAPHS 500
#define HIDDEN   128
#define NUM_TASKS 12

#define USE_TDM 1   // tensor_load_to_lds for LDS weight staging

// ---------------------------------------------------------------------------
// WMMA types (CDNA5 / gfx1250, wave32)
// ---------------------------------------------------------------------------
typedef __attribute__((ext_vector_type(16))) _Float16 v16h;
typedef __attribute__((ext_vector_type(8)))  _Float16 v8h;
typedef __attribute__((ext_vector_type(8)))  float    v8f;
typedef __attribute__((ext_vector_type(4)))  unsigned int u32x4;
typedef __attribute__((ext_vector_type(8)))  unsigned int u32x8;

union V16 { v16h v; v8h h[2]; };
static_assert(sizeof(V16) == 32, "V16 must be 32B");

// ---------------------------------------------------------------------------
// TDM: async DMA of a contiguous 32KB f16 weight image (128x128) into LDS.
// D# per CDNA5 ISA:
//  group0: [1:0]=count=1, [63:32]=lds_addr, [120:64]=global_addr, [127:126]=2
//  group1: data_size=1 (2B) at [17:16]; tensor_dim0 [79:48]=16384;
//          tensor_dim1 [111:80]=1; tile_dim0 [127:112]=16384;
//          tile_dim1 [143:128]=1; tensor_dim0_stride [207:160]=16384
// 2-operand form => VADDR2/VADDR3 = NULL (2D tensor). Tracked by TENSORcnt.
// ---------------------------------------------------------------------------
__device__ inline void tdm_load_w32k(const _Float16* gsrc, unsigned lds_byte_off) {
#if USE_TDM
    const unsigned long long ga = (unsigned long long)(__SIZE_TYPE__)gsrc;
    u32x4 g0;
    g0[0] = 1u;                                             // count=1 (valid), user mode
    g0[1] = lds_byte_off;                                   // lds_addr (wave-relative bytes)
    g0[2] = (unsigned)(ga & 0xFFFFFFFFu);                   // global_addr[31:0]
    g0[3] = (unsigned)((ga >> 32) & 0x01FFFFFFu) | 0x80000000u; // addr[56:32] | type=2
    u32x8 g1;
    g1[0] = 1u << 16;        // workgroup_mask=0, data_size=1 (2 bytes), no flags/pad
    g1[1] = 0x40000000u;     // tensor_dim0 low16 (=16384) in [31:16]
    g1[2] = 1u << 16;        // tensor_dim0 hi16=0 | tensor_dim1 low16=1
    g1[3] = 0x40000000u;     // tensor_dim1 hi16=0 | tile_dim0=16384
    g1[4] = 1u;              // tile_dim1=1, tile_dim2=0
    g1[5] = 16384u;          // tensor_dim0_stride low32 (elements)
    g1[6] = 0u;              // stride hi16 | tensor_dim1_stride low16
    g1[7] = 0u;              // tensor_dim1_stride hi32
    asm volatile("tensor_load_to_lds %0, %1" :: "s"(g0), "s"(g1) : "memory");
#endif
}

// ---------------------------------------------------------------------------
// Generic fill
// ---------------------------------------------------------------------------
__global__ __launch_bounds__(256) void fill_kernel(float* __restrict__ p, float v, int n) {
    int i = blockIdx.x * 256 + threadIdx.x;
    if (i < n) p[i] = v;
}

// ---------------------------------------------------------------------------
// Weight prep: W1T/W2T[n*128+k] = (f16) W[k*128+n]   (transpose + f32->f16)
// ---------------------------------------------------------------------------
__global__ __launch_bounds__(256) void wprep_kernel(const float* __restrict__ w1,
                                                    const float* __restrict__ w2,
                                                    _Float16* __restrict__ w1t,
                                                    _Float16* __restrict__ w2t) {
    int i = blockIdx.x * 256 + threadIdx.x;           // 0..16383
    if (i >= HIDDEN * HIDDEN) return;
    int n = i >> 7, k = i & 127;
    w1t[i] = (_Float16)w1[k * HIDDEN + n];
    w2t[i] = (_Float16)w2[k * HIDDEN + n];
}

// ---------------------------------------------------------------------------
// Edge-embedding table: E16[h][c] = ee[h,:8] @ elw[:, c] + elb[c], h in [0,16)
// ---------------------------------------------------------------------------
__global__ __launch_bounds__(256) void e16_kernel(const float* __restrict__ ee,
                                                  const float* __restrict__ elw,
                                                  const float* __restrict__ elb,
                                                  float* __restrict__ E16) {
    int i = blockIdx.x * 256 + threadIdx.x;           // 0..2047
    if (i >= 16 * HIDDEN) return;
    int h = i >> 7, c = i & 127;
    float s = elb[c];
#pragma unroll
    for (int k = 0; k < 8; ++k) s = fmaf(ee[h * 8 + k], elw[k * HIDDEN + c], s);
    E16[i] = s;
}

// ---------------------------------------------------------------------------
// Node embedding gather + 100->128 projection. One 128-thread block per node.
// ---------------------------------------------------------------------------
__global__ __launch_bounds__(128) void proj_kernel(const int* __restrict__ x,
                                                   const float* __restrict__ e0,
                                                   const float* __restrict__ e1,
                                                   const float* __restrict__ e2,
                                                   const float* __restrict__ e3,
                                                   const float* __restrict__ e4,
                                                   const float* __restrict__ pw,
                                                   const float* __restrict__ pb,
                                                   float* __restrict__ out) {
    __shared__ float f[100];
    const int n = blockIdx.x;
    const int t = threadIdx.x;
    if (t < 100) {
        const float* e; int off, dim, voc, col;
        if      (t < 64) { e = e0; off = 0;  dim = 64; voc = 120; col = 0; }
        else if (t < 80) { e = e1; off = 64; dim = 16; voc = 10;  col = 1; }
        else if (t < 88) { e = e2; off = 80; dim = 8;  voc = 7;   col = 2; }
        else if (t < 96) { e = e3; off = 88; dim = 8;  voc = 5;   col = 3; }
        else             { e = e4; off = 96; dim = 4;  voc = 2;   col = 4; }
        int idx = x[n * 5 + col];
        idx = idx < 0 ? 0 : (idx > voc - 1 ? voc - 1 : idx);
        f[t] = e[idx * dim + (t - off)];
    }
    __syncthreads();
    float s = pb[t];
    for (int k = 0; k < 100; ++k) s = fmaf(f[k], pw[k * HIDDEN + t], s);
    out[n * HIDDEN + t] = s;
}

// ---------------------------------------------------------------------------
// Edge scatter: one wave per edge; aggr[dst] += relu(h[src] + E16[hash])
// ---------------------------------------------------------------------------
__global__ __launch_bounds__(256) void edge_scatter_kernel(const int* __restrict__ ei,
                                                           const int* __restrict__ ea,
                                                           const float* __restrict__ h,
                                                           const float* __restrict__ E16,
                                                           float* __restrict__ aggr) {
    const int e    = (blockIdx.x * 256 + threadIdx.x) >> 5;
    const int lane = threadIdx.x & 31;
    if (e >= N_EDGES) return;
    const int src = ei[e];
    const int dst = ei[N_EDGES + e];
    const int hh  = (ea[e * 3 + 0] + 3 * ea[e * 3 + 1] + 7 * ea[e * 3 + 2]) & 15;
    const float4 xv = ((const float4*)(h + (size_t)src * HIDDEN))[lane];
    const float4 ev = ((const float4*)(E16 + hh * HIDDEN))[lane];
    float4 m;
    m.x = fmaxf(xv.x + ev.x, 0.f);
    m.y = fmaxf(xv.y + ev.y, 0.f);
    m.z = fmaxf(xv.z + ev.z, 0.f);
    m.w = fmaxf(xv.w + ev.w, 0.f);
    float* dp = aggr + (size_t)dst * HIDDEN + lane * 4;
    atomicAdd(dp + 0, m.x);
    atomicAdd(dp + 1, m.y);
    atomicAdd(dp + 2, m.z);
    atomicAdd(dp + 3, m.w);
}

// ---------------------------------------------------------------------------
// Fused GINE node update, double WMMA GEMM, TDM weight staging:
//   A  = (1+eps)*x + aggr                  (16 rows per wave, staged f16 in LDS)
//   T  = relu(A @ W1 + b1)                 (WMMA f32_16x16x32_f16, f32 accum)
//   O  = T @ W2 + b2 ; O = relu(O); optional eval-BN scale/shift
// Block = 128 threads (4 waves) -> 64 rows. LDS = 32KB weights + 16KB A = 48KB.
// Weights arrive via tensor_load_to_lds (wave 0 issues, TENSORcnt fences),
// overlapped with per-wave A staging / epilogue-1 LDS writes.
// ---------------------------------------------------------------------------
__global__ __launch_bounds__(128) void mlp_gine_kernel(const float* __restrict__ x,
                                                       const float* __restrict__ aggr,
                                                       const float* __restrict__ epsp,
                                                       const _Float16* __restrict__ W1T,
                                                       const _Float16* __restrict__ W2T,
                                                       const float* __restrict__ b1,
                                                       const float* __restrict__ b2,
                                                       const float* __restrict__ bn_g,
                                                       const float* __restrict__ bn_b,
                                                       float* __restrict__ out) {
    extern __shared__ char smem_raw[];
    _Float16* sW = (_Float16*)smem_raw;               // 128x128 f16 = 32KB @ offset 0
    _Float16* sA = (_Float16*)(smem_raw + 32768);     // 4 waves * 16x128 f16 = 16KB

    const int tid  = threadIdx.x;
    const int lane = tid & 31;
    const int wid  = tid >> 5;
    const int lm   = lane & 15;       // N / M-row selector within half-wave
    const int g    = lane >> 4;       // lane group (K staggering)
    const int r0   = blockIdx.x * 64 + wid * 16;
    _Float16* sAw  = sA + wid * 2048;

#if USE_TDM
    // --- async: wave 0 kicks the W1^T DMA into LDS; overlaps A staging ---
    if (wid == 0) tdm_load_w32k(W1T, 0u);
#else
    {
        const uint4* s = (const uint4*)W1T;
        uint4*       d = (uint4*)sW;
#pragma unroll
        for (int i = 0; i < 16; ++i) d[tid + i * 128] = s[tid + i * 128];
    }
#endif

    // --- stage A = (1+eps)*x + aggr for this wave's 16 rows, f32 -> f16 ---
    const float epsv = 1.0f + epsp[0];
    for (int i = lane; i < 2048; i += 32) {
        int rr = i >> 7, cc = i & 127;
        int row = r0 + rr;
        if (row >= N_NODES) row = 0;                 // clamped garbage, stores guarded
        float v = fmaf(epsv, x[(size_t)row * HIDDEN + cc], aggr[(size_t)row * HIDDEN + cc]);
        sAw[i] = (_Float16)v;
    }
#if USE_TDM
    if (wid == 0) __builtin_amdgcn_s_wait_tensorcnt(0);  // W1 resident before barrier
#endif
    __syncthreads();

    const v8f vzero = {};
    v8f acc[8];
#pragma unroll
    for (int nt = 0; nt < 8; ++nt) acc[nt] = vzero;

    // --- GEMM1: 4 k-steps x 8 n-tiles ---
#pragma unroll
    for (int ks = 0; ks < 4; ++ks) {
        V16 a;
        const _Float16* ap = sAw + lm * 128 + ks * 32 + g * 8;
        a.h[0] = *(const v8h*)(ap);
        a.h[1] = *(const v8h*)(ap + 16);
#pragma unroll
        for (int nt = 0; nt < 8; ++nt) {
            V16 bu; const _Float16* bp = sW + (nt * 16 + lm) * 128 + ks * 32 + g * 8;
            bu.h[0] = *(const v8h*)(bp); bu.h[1] = *(const v8h*)(bp + 16);
            v16h b = bu.v;
            acc[nt] = __builtin_amdgcn_wmma_f32_16x16x32_f16(
                false, a.v, false, b, (short)0, acc[nt], false, false);
        }
    }
    __syncthreads();   // everyone done reading sW/sA of GEMM1

#if USE_TDM
    // --- async: W2^T DMA; overlaps epilogue-1 writes into disjoint sA ---
    if (wid == 0) tdm_load_w32k(W2T, 0u);
#endif

    // --- epilogue 1: relu(acc + b1) -> f16 back into this wave's A tile ---
#pragma unroll
    for (int nt = 0; nt < 8; ++nt) {
        const int n = nt * 16 + lm;
        const float bias = b1[n];
#pragma unroll
        for (int r = 0; r < 8; ++r) {
            float v = acc[nt][r] + bias;
            v = v > 0.f ? v : 0.f;
            sAw[(r + 8 * g) * 128 + n] = (_Float16)v;
        }
    }

#if USE_TDM
    if (wid == 0) __builtin_amdgcn_s_wait_tensorcnt(0);  // W2 resident before barrier
#else
    {
        const uint4* s = (const uint4*)W2T;
        uint4*       d = (uint4*)sW;
#pragma unroll
        for (int i = 0; i < 16; ++i) d[tid + i * 128] = s[tid + i * 128];
    }
#endif
    __syncthreads();

#pragma unroll
    for (int nt = 0; nt < 8; ++nt) acc[nt] = vzero;

    // --- GEMM2 ---
#pragma unroll
    for (int ks = 0; ks < 4; ++ks) {
        V16 a;
        const _Float16* ap = sAw + lm * 128 + ks * 32 + g * 8;
        a.h[0] = *(const v8h*)(ap);
        a.h[1] = *(const v8h*)(ap + 16);
#pragma unroll
        for (int nt = 0; nt < 8; ++nt) {
            V16 bu; const _Float16* bp = sW + (nt * 16 + lm) * 128 + ks * 32 + g * 8;
            bu.h[0] = *(const v8h*)(bp); bu.h[1] = *(const v8h*)(bp + 16);
            v16h b = bu.v;
            acc[nt] = __builtin_amdgcn_wmma_f32_16x16x32_f16(
                false, a.v, false, b, (short)0, acc[nt], false, false);
        }
    }

    // --- epilogue 2: +b2, relu, optional eval-BN, guarded store ---
    const bool useBN = (bn_g != nullptr);
    const float inv = rsqrtf(1.0f + 1e-5f);
#pragma unroll
    for (int nt = 0; nt < 8; ++nt) {
        const int n = nt * 16 + lm;
        const float bias = b2[n];
        float sc = 1.f, sh = 0.f;
        if (useBN) { sc = bn_g[n] * inv; sh = bn_b[n]; }
#pragma unroll
        for (int r = 0; r < 8; ++r) {
            const int row = r0 + r + 8 * g;
            float v = acc[nt][r] + bias;
            v = v > 0.f ? v : 0.f;
            v = fmaf(v, sc, sh);
            if (row < N_NODES) out[(size_t)row * HIDDEN + n] = v;
        }
    }
}

// ---------------------------------------------------------------------------
// Pooling: gate, segment-max, softmax, weighted segment-sum, head
// ---------------------------------------------------------------------------
__device__ inline void atomicMaxFloat(float* addr, float val) {
    unsigned int* ua = (unsigned int*)addr;
    unsigned int old = *ua;
    while (__uint_as_float(old) < val) {
        unsigned int assumed = old;
        old = atomicCAS(ua, assumed, __float_as_uint(val));
        if (old == assumed) break;
    }
}

__global__ __launch_bounds__(256) void gate_kernel(const float* __restrict__ h,
                                                   const float* __restrict__ gw,
                                                   const float* __restrict__ gb,
                                                   const int* __restrict__ batch,
                                                   float* __restrict__ gate,
                                                   float* __restrict__ m) {
    const int node = (blockIdx.x * 256 + threadIdx.x) >> 5;
    const int lane = threadIdx.x & 31;
    if (node >= N_NODES) return;
    const float4 hv = ((const float4*)(h + (size_t)node * HIDDEN))[lane];
    const float4 wv = ((const float4*)gw)[lane];
    float s = hv.x * wv.x + hv.y * wv.y + hv.z * wv.z + hv.w * wv.w;
    for (int off = 16; off > 0; off >>= 1) s += __shfl_xor(s, off, 32);
    if (lane == 0) {
        s += gb[0];
        gate[node] = s;
        atomicMaxFloat(&m[batch[node]], s);
    }
}

__global__ __launch_bounds__(256) void softmax_num_kernel(const float* __restrict__ gate,
                                                          const int* __restrict__ batch,
                                                          const float* __restrict__ m,
                                                          float* __restrict__ exv,
                                                          float* __restrict__ den) {
    const int n = blockIdx.x * 256 + threadIdx.x;
    if (n >= N_NODES) return;
    const int b = batch[n];
    const float e = __expf(gate[n] - m[b]);
    exv[n] = e;
    atomicAdd(&den[b], e);
}

__global__ __launch_bounds__(256) void pool_kernel(const float* __restrict__ h,
                                                   const float* __restrict__ exv,
                                                   const float* __restrict__ den,
                                                   const int* __restrict__ batch,
                                                   float* __restrict__ hg) {
    const int i = blockIdx.x * 256 + threadIdx.x;
    if (i >= N_NODES * HIDDEN) return;
    const int n = i >> 7, c = i & 127;
    const int b = batch[n];
    const float coef = exv[n] / den[b];
    atomicAdd(&hg[b * HIDDEN + c], coef * h[i]);
}

__global__ __launch_bounds__(128) void head_kernel(const float* __restrict__ hg,
                                                   const float* __restrict__ w1,
                                                   const float* __restrict__ bb1,
                                                   const float* __restrict__ w2,
                                                   const float* __restrict__ bb2,
                                                   float* __restrict__ out) {
    __shared__ float row[HIDDEN];
    __shared__ float t1[HIDDEN];
    const int gph = blockIdx.x;
    const int t = threadIdx.x;
    row[t] = hg[gph * HIDDEN + t];
    __syncthreads();
    float s = bb1[t];
    for (int k = 0; k < HIDDEN; ++k) s = fmaf(row[k], w1[k * HIDDEN + t], s);
    t1[t] = fmaxf(s, 0.f);
    __syncthreads();
    if (t < NUM_TASKS) {
        float s2 = bb2[t];
        for (int k = 0; k < HIDDEN; ++k) s2 = fmaf(t1[k], w2[k * NUM_TASKS + t], s2);
        out[gph * NUM_TASKS + t] = s2;
    }
}

// ---------------------------------------------------------------------------
// Orchestration
// ---------------------------------------------------------------------------
extern "C" void kernel_launch(void* const* d_in, const int* in_sizes, int n_in,
                              void* d_out, int out_size, void* d_ws, size_t ws_size,
                              hipStream_t stream) {
    (void)in_sizes; (void)n_in; (void)out_size; (void)ws_size;

    // --- inputs (setup_inputs dict order, params flattened in insertion order) ---
    const int* x   = (const int*)d_in[0];
    const int* ei  = (const int*)d_in[1];
    const int* ea  = (const int*)d_in[2];
    const int* bat = (const int*)d_in[3];
    const float* embs[5] = {(const float*)d_in[4], (const float*)d_in[5],
                            (const float*)d_in[6], (const float*)d_in[7],
                            (const float*)d_in[8]};
    const float* proj_w = (const float*)d_in[9];
    const float* proj_b = (const float*)d_in[10];
    // g1: 11..18, g2: 19..26, g3: 27..34  (ee, elw, elb, eps, m1w, m1b, m2w, m2b)
    const float* bn_g_arr[3] = {(const float*)d_in[35], (const float*)d_in[37], nullptr};
    const float* bn_b_arr[3] = {(const float*)d_in[36], (const float*)d_in[38], nullptr};
    const float* gate_w = (const float*)d_in[39];
    const float* gate_b = (const float*)d_in[40];
    const float* h1w = (const float*)d_in[41];
    const float* h1b = (const float*)d_in[42];
    const float* h2w = (const float*)d_in[43];
    const float* h2b = (const float*)d_in[44];

    // --- workspace layout (bytes) ---
    char* ws = (char*)d_ws;
    float*    h    = (float*)(ws + 0);                       // 50000*128*4 = 25,600,000
    float*    aggr = (float*)(ws + 25600000);                // 25,600,000
    _Float16* W1T  = (_Float16*)(ws + 51200000);             // 32,768
    _Float16* W2T  = (_Float16*)(ws + 51232768);             // 32,768
    float*    E16  = (float*)(ws + 51265536);                // 8,192
    float*    gate = (float*)(ws + 51273728);                // 200,000
    float*    exv  = (float*)(ws + 51473728);                // 200,000
    float*    m    = (float*)(ws + 51673728);                // 2,000
    float*    den  = (float*)(ws + 51675728);                // 2,000
    float*    hg   = (float*)(ws + 51677728);                // 256,000

    // --- embedding + projection -> h ---
    proj_kernel<<<N_NODES, 128, 0, stream>>>(x, embs[0], embs[1], embs[2], embs[3],
                                             embs[4], proj_w, proj_b, h);

    // --- 3 GINE blocks (in-place on h) ---
    for (int b = 0; b < 3; ++b) {
        const int base = 11 + 8 * b;
        const float* ee   = (const float*)d_in[base + 0];
        const float* elw  = (const float*)d_in[base + 1];
        const float* elb  = (const float*)d_in[base + 2];
        const float* epsp = (const float*)d_in[base + 3];
        const float* m1w  = (const float*)d_in[base + 4];
        const float* m1b  = (const float*)d_in[base + 5];
        const float* m2w  = (const float*)d_in[base + 6];
        const float* m2b  = (const float*)d_in[base + 7];

        wprep_kernel<<<64, 256, 0, stream>>>(m1w, m2w, W1T, W2T);
        e16_kernel<<<8, 256, 0, stream>>>(ee, elw, elb, E16);
        fill_kernel<<<(N_NODES * HIDDEN + 255) / 256, 256, 0, stream>>>(aggr, 0.f, N_NODES * HIDDEN);
        edge_scatter_kernel<<<(N_EDGES * 32 + 255) / 256, 256, 0, stream>>>(ei, ea, h, E16, aggr);
        mlp_gine_kernel<<<(N_NODES + 63) / 64, 128, 49152, stream>>>(
            h, aggr, epsp, W1T, W2T, m1b, m2b, bn_g_arr[b], bn_b_arr[b], h);
    }

    // --- attention pooling ---
    fill_kernel<<<(N_GRAPHS + 255) / 256, 256, 0, stream>>>(m, -3.0e38f, N_GRAPHS);
    fill_kernel<<<(N_GRAPHS + 255) / 256, 256, 0, stream>>>(den, 0.f, N_GRAPHS);
    fill_kernel<<<(N_GRAPHS * HIDDEN + 255) / 256, 256, 0, stream>>>(hg, 0.f, N_GRAPHS * HIDDEN);
    gate_kernel<<<(N_NODES * 32 + 255) / 256, 256, 0, stream>>>(h, gate_w, gate_b, bat, gate, m);
    softmax_num_kernel<<<(N_NODES + 255) / 256, 256, 0, stream>>>(gate, bat, m, exv, den);
    pool_kernel<<<(N_NODES * HIDDEN + 255) / 256, 256, 0, stream>>>(h, exv, den, bat, hg);

    // --- head -> d_out [500,12] ---
    head_kernel<<<N_GRAPHS, 128, 0, stream>>>(hg, h1w, h1b, h2w, h2b, (float*)d_out);
}
